// MSDeformableAttention3DLite_62517543960683
// MI455X (gfx1250) — compile-verified, matrix operand order checked by reading, not run on hardware
//
#include <hip/hip_runtime.h>

typedef __attribute__((ext_vector_type(16))) __bf16 v16bf;
typedef __attribute__((ext_vector_type(8)))  __bf16 v8bf;
typedef __attribute__((ext_vector_type(8)))  float  v8f;

__device__ __forceinline__ unsigned short f32_to_bf16_rne(float f) {
  unsigned u = __builtin_bit_cast(unsigned, f);
  return (unsigned short)((u + 0x7FFFu + ((u >> 16) & 1u)) >> 16);
}

// Build Wt[N][K] bf16 from up to two row-major f32 weights W0[K][split], W1[K][N-split].
__global__ void pack_wt_kernel(const float* __restrict__ w0,
                               const float* __restrict__ w1,
                               int split, int N, int K,
                               unsigned short* __restrict__ out) {
  int idx = blockIdx.x * blockDim.x + threadIdx.x;
  if (idx >= N * K) return;
  int n = idx / K;
  int k = idx - n * K;
  float f = (n < split) ? w0[(size_t)k * split + n]
                        : w1[(size_t)k * (N - split) + (n - split)];
  out[idx] = f32_to_bf16_rne(f);
}

// C[M,N] = A[M,256](f32) * W[256,N] + bias, W given as Wt[N][256] bf16.
// Workgroup: 16 M-rows x 128 N-cols, 8 waves, each wave one 16x16 WMMA tile.
__global__ __launch_bounds__(256)
void gemm_bf16_wmma_kernel(const float* __restrict__ A,
                           const unsigned short* __restrict__ Wt,
                           const float* __restrict__ bias0,
                           const float* __restrict__ bias1,
                           int split, int M, int N,
                           float* __restrict__ C) {
  __shared__ unsigned short sA[16 * 264];   // 16 rows x 256 bf16, +8 pad per row
  const int tid  = threadIdx.x;
  const int wave = tid >> 5;
  const int lane = tid & 31;
  const int m0   = blockIdx.x * 16;

  // cooperative A tile load: f32 -> bf16 into LDS
  for (int i = tid; i < 16 * 256; i += 256) {
    int r = i >> 8, c = i & 255;
    sA[r * 264 + c] = f32_to_bf16_rne(A[(size_t)(m0 + r) * 256 + c]);
  }
  __syncthreads();

  const int n0 = blockIdx.y * 128 + wave * 16;
  if (n0 >= N) return;

  const int half = lane >> 4;       // 0: k in {0..7,16..23}; 1: {8..15,24..31}
  const int sub  = lane & 15;       // M row (A) / N col (B)
  const int kb   = half << 3;

  const unsigned short* wrow = Wt + (size_t)(n0 + sub) * 256 + kb;
  const unsigned short* arow = sA + sub * 264 + kb;

  v8f acc = {};
#pragma unroll
  for (int kk = 0; kk < 256; kk += 32) {
    v8bf alo = *(const v8bf*)(arow + kk);
    v8bf ahi = *(const v8bf*)(arow + kk + 16);
    v8bf blo = *(const v8bf*)(wrow + kk);
    v8bf bhi = *(const v8bf*)(wrow + kk + 16);
    v16bf af = __builtin_shufflevector(alo, ahi, 0,1,2,3,4,5,6,7,8,9,10,11,12,13,14,15);
    v16bf bf = __builtin_shufflevector(blo, bhi, 0,1,2,3,4,5,6,7,8,9,10,11,12,13,14,15);
    acc = __builtin_amdgcn_wmma_f32_16x16x32_bf16(false, af, false, bf,
                                                  (short)0, acc, false, false);
  }

  const int cn = n0 + sub;
  const float bv = (cn < split) ? bias0[cn] : bias1[cn - split];
  const int mb = m0 + half * 8;
#pragma unroll
  for (int r = 0; r < 8; ++r) {
    int m = mb + r;
    if (m < M) C[(size_t)m * N + cn] = acc[r] + bv;   // D layout: vgpr r -> M=mb+r, N=cn
  }
}

// One wave per (b, q, head). Phase 1: lane = sample point (L*P = 32), softmax +
// location computation. Phase 2: lane = channel d (D = 32), coalesced bilinear gather.
__global__ __launch_bounds__(256)
void ms_deform_sample_kernel(const float* __restrict__ qp,    // [B*Q, 768]
                             const float* __restrict__ vproj, // [B*S, 256]
                             const float* __restrict__ refp,  // [B, Q, 4, 2]
                             const int*   __restrict__ ss,    // [4, 2]
                             float* __restrict__ out,         // [B, Q, 256]
                             int BQ, int Q) {
  const int wid  = blockIdx.x * (blockDim.x >> 5) + (threadIdx.x >> 5);
  const int lane = threadIdx.x & 31;
  if (wid >= BQ * 8) return;
  const int h  = wid & 7;
  const int bq = wid >> 3;
  const int b  = bq / Q;

  const int H0 = ss[0], W0 = ss[1], H1 = ss[2], W1 = ss[3];
  const int H2 = ss[4], W2 = ss[5], H3 = ss[6], W3 = ss[7];
  const int a0 = H0 * W0, a1 = H1 * W1, a2 = H2 * W2;
  const int S  = a0 + a1 + a2 + H3 * W3;

  const int l = lane >> 3, p = lane & 7;
  const int Hl = (l == 0) ? H0 : (l == 1) ? H1 : (l == 2) ? H2 : H3;
  const int Wl = (l == 0) ? W0 : (l == 1) ? W1 : (l == 2) ? W2 : W3;
  const int lb = (l == 0) ? 0 : (l == 1) ? a0 : (l == 2) ? a0 + a1 : a0 + a1 + a2;

  const float* qrow = qp + (size_t)bq * 768;

  // softmax over all 32 (level, point) logits of this head
  float logit = qrow[512 + h * 32 + lane];
  float mx = logit;
  for (int o = 16; o > 0; o >>= 1) mx = fmaxf(mx, __shfl_xor(mx, o, 32));
  float e = __expf(logit - mx);
  float sum = e;
  for (int o = 16; o > 0; o >>= 1) sum += __shfl_xor(sum, o, 32);
  const float wgt = e / sum;

  // sampling location (px = refx*W + offx - 0.5 ; refz = p % 4)
  const float ox = qrow[h * 64 + lane * 2 + 0];
  const float oy = qrow[h * 64 + lane * 2 + 1];
  const float* rp = refp + ((size_t)bq * 4 + (p & 3)) * 2;
  const float px = rp[0] * (float)Wl + ox - 0.5f;
  const float py = rp[1] * (float)Hl + oy - 0.5f;

  // gather phase: lane = channel
  const float* vb = vproj + (size_t)b * S * 256 + h * 32 + lane;
  float acc = 0.f;
#pragma unroll 4
  for (int j = 0; j < 32; ++j) {
    const float pxj = __shfl(px, j, 32);
    const float pyj = __shfl(py, j, 32);
    const float wj  = __shfl(wgt, j, 32);
    const int   Wj  = __shfl(Wl, j, 32);
    const int   Hj  = __shfl(Hl, j, 32);
    const int   bj  = __shfl(lb, j, 32);

    const float fx = floorf(pxj), fy = floorf(pyj);
    const float tx = pxj - fx, ty = pyj - fy;
    const int x0 = (int)fx, y0 = (int)fy, x1 = x0 + 1, y1 = y0 + 1;

    const float mx0 = (x0 >= 0 && x0 < Wj) ? 1.f : 0.f;
    const float mx1 = (x1 >= 0 && x1 < Wj) ? 1.f : 0.f;
    const float my0 = (y0 >= 0 && y0 < Hj) ? 1.f : 0.f;
    const float my1 = (y1 >= 0 && y1 < Hj) ? 1.f : 0.f;
    const int xc0 = min(max(x0, 0), Wj - 1), yc0 = min(max(y0, 0), Hj - 1);
    const int xc1 = min(max(x1, 0), Wj - 1), yc1 = min(max(y1, 0), Hj - 1);

    const float v00 = vb[(size_t)(bj + yc0 * Wj + xc0) * 256] * (mx0 * my0);
    const float v01 = vb[(size_t)(bj + yc0 * Wj + xc1) * 256] * (mx1 * my0);
    const float v10 = vb[(size_t)(bj + yc1 * Wj + xc0) * 256] * (mx0 * my1);
    const float v11 = vb[(size_t)(bj + yc1 * Wj + xc1) * 256] * (mx1 * my1);

    acc += wj * ((v00 * (1.f - tx) + v01 * tx) * (1.f - ty) +
                 (v10 * (1.f - tx) + v11 * tx) * ty);
  }
  out[(size_t)bq * 256 + h * 32 + lane] = acc;
}

extern "C" void kernel_launch(void* const* d_in, const int* in_sizes, int n_in,
                              void* d_out, int out_size, void* d_ws, size_t ws_size,
                              hipStream_t stream) {
  (void)n_in; (void)out_size; (void)ws_size;
  const float* query  = (const float*)d_in[0];
  const float* value  = (const float*)d_in[1];
  const float* refp   = (const float*)d_in[2];
  const float* w_off  = (const float*)d_in[3];
  const float* b_off  = (const float*)d_in[4];
  const float* w_attn = (const float*)d_in[5];
  const float* b_attn = (const float*)d_in[6];
  const float* w_val  = (const float*)d_in[7];
  const float* b_val  = (const float*)d_in[8];
  const int*   ss     = (const int*)d_in[9];

  const int E = 256;
  const int M_v = in_sizes[1] / E;   // B*S = 78240
  const int M_q = in_sizes[0] / E;   // B*Q = 40000
  const int Q = 10000;
  const int BQ = M_q;

  // workspace layout
  char* ws = (char*)d_ws;
  unsigned short* WtVal = (unsigned short*)ws;                       // 256*256*2
  unsigned short* WtQ   = (unsigned short*)(ws + 131072);            // 768*256*2
  float* vproj          = (float*)(ws + 131072 + 393216);            // M_v*256*4
  float* qproj          = (float*)(ws + 524288 + (size_t)M_v * 256 * 4);

  // 1) pack weights to bf16 Wt[N][K]
  pack_wt_kernel<<<(256 * 256 + 255) / 256, 256, 0, stream>>>(
      w_val, w_val, 256, 256, 256, WtVal);
  pack_wt_kernel<<<(768 * 256 + 255) / 256, 256, 0, stream>>>(
      w_off, w_attn, 512, 768, 256, WtQ);

  // 2) value projection: [M_v,256] x [256,256]
  {
    dim3 grid((M_v + 15) / 16, 256 / 128);
    gemm_bf16_wmma_kernel<<<grid, 256, 0, stream>>>(
        value, WtVal, b_val, b_val, 256, M_v, 256, vproj);
  }
  // 3) query projection (offsets || attn logits): [M_q,256] x [256,768]
  {
    dim3 grid((M_q + 15) / 16, 768 / 128);
    gemm_bf16_wmma_kernel<<<grid, 256, 0, stream>>>(
        query, WtQ, b_off, b_attn, 512, M_q, 768, qproj);
  }
  // 4) softmax + bilinear sampling + weighted reduce
  {
    int waves = BQ * 8;                      // one wave per (b,q,head)
    int blocks = (waves + 7) / 8;            // 8 waves per 256-thread block
    ms_deform_sample_kernel<<<blocks, 256, 0, stream>>>(
        qproj, vproj, refp, ss, (float*)d_out, BQ, Q);
  }
}
